// NodeUpdate_57629871177758
// MI455X (gfx1250) — compile-verified, matrix operand order checked by reading, not run on hardware
//
#include <hip/hip_runtime.h>
#include <hip/hip_bf16.h>

// ---------------------------------------------------------------------------
// Problem constants (match reference)
// ---------------------------------------------------------------------------
#define N_NODES 50000
#define N_EDGES 800000
#define EC_DIM  88
#define NC      32
#define HC      256
#define OC      32
#define IN_DIM  472          // NC + 5*EC_DIM
#define K1_PAD  480          // IN_DIM padded to multiple of 32 for bf16 WMMA

typedef unsigned long long ull;
typedef __attribute__((ext_vector_type(8)))  unsigned short v8us;
typedef __attribute__((ext_vector_type(16))) unsigned short v16us;
typedef __attribute__((ext_vector_type(16))) __bf16         v16bf;
typedef __attribute__((ext_vector_type(8)))  float          v8f;

// ---------------------------------------------------------------------------
// Helpers
// ---------------------------------------------------------------------------
// Monotonic (order-preserving) uint encoding of float for atomicMin/Max.
__device__ __forceinline__ unsigned f2ord(float f) {
    unsigned u = __float_as_uint(f);
    return (u & 0x80000000u) ? ~u : (u | 0x80000000u);
}
__device__ __forceinline__ float ord2f(unsigned u) {
    unsigned x = (u & 0x80000000u) ? (u ^ 0x80000000u) : ~u;
    return __uint_as_float(x);
}
// float -> bf16 bits, round-to-nearest-even
__device__ __forceinline__ unsigned short f2bf(float f) {
    unsigned u = __float_as_uint(f);
    u += 0x7FFFu + ((u >> 16) & 1u);
    return (unsigned short)(u >> 16);
}
__device__ __forceinline__ void store_out(float* p, float v)          { *p = v; }
__device__ __forceinline__ void store_out(unsigned short* p, float v) { *p = f2bf(v); }

__device__ __forceinline__ v16bf make_frag(v8us lo, v8us hi) {
    v16us v = __builtin_shufflevector(lo, hi, 0,1,2,3,4,5,6,7,8,9,10,11,12,13,14,15);
    return __builtin_bit_cast(v16bf, v);
}

// ---------------------------------------------------------------------------
// Kernel 1: init aggregation buffers
// ---------------------------------------------------------------------------
__global__ void init_agg(unsigned* __restrict__ umin, unsigned* __restrict__ umax,
                         float* __restrict__ fsum, float* __restrict__ fcnt,
                         ull* __restrict__ pmin, ull* __restrict__ pmax) {
    long long gid = (long long)blockIdx.x * blockDim.x + threadIdx.x;
    if (gid < (long long)N_NODES * EC_DIM) {
        umin[gid] = 0xFFFFFFFFu;   // +inf in ordered encoding
        umax[gid] = 0u;            // -inf in ordered encoding
        fsum[gid] = 0.0f;
    }
    if (gid < N_NODES) {
        fcnt[gid] = 0.0f;
        pmin[gid] = ~0ull;         // (key=+inf, eid=MAX) -> clamps to E-1 if untouched
        pmax[gid] = 0ull;          // (key=-inf, ~eid=0 -> eid=MAX)
    }
}

// ---------------------------------------------------------------------------
// Kernel 2: single-pass edge segment reduction (min/max/sum/count/arg*)
// One thread per (edge, feature); coalesced edge_attr read.
// ---------------------------------------------------------------------------
__global__ void edge_agg(const float* __restrict__ ea, const int* __restrict__ eidx,
                         unsigned* __restrict__ umin, unsigned* __restrict__ umax,
                         float* __restrict__ fsum, float* __restrict__ fcnt,
                         ull* __restrict__ pmin, ull* __restrict__ pmax) {
    long long gid = (long long)blockIdx.x * blockDim.x + threadIdx.x;
    if (gid >= (long long)N_EDGES * EC_DIM) return;
    int e = (int)(gid / EC_DIM);
    int f = (int)(gid - (long long)e * EC_DIM);
    int dest = eidx[N_EDGES + e];          // edge_index[1][e]
    float v = ea[gid];
    unsigned o = f2ord(v);
    size_t off = (size_t)dest * EC_DIM + f;
    atomicMin(&umin[off], o);
    atomicMax(&umax[off], o);
    atomicAdd(&fsum[off], v);
    if (f == 0) {
        atomicAdd(&fcnt[dest], 1.0f);
        // argmax: max key, tie -> smallest eid (maximize ~eid)
        atomicMax(&pmax[dest], ((ull)o << 32) | (unsigned)(0xFFFFFFFFu - (unsigned)e));
        // argmin: min key, tie -> smallest eid
        atomicMin(&pmin[dest], ((ull)o << 32) | (unsigned)e);
    }
}

// ---------------------------------------------------------------------------
// Kernel 3: build concatenated input matrix [N_NODES, 480] in bf16
// layout: [x(32) | ea[argmin](88) | ea[argmax](88) | min(88) | mean(88) | max(88) | pad(8)]
// ---------------------------------------------------------------------------
__global__ void build_inputs(const float* __restrict__ x, const float* __restrict__ ea,
                             const unsigned* __restrict__ umin, const unsigned* __restrict__ umax,
                             const float* __restrict__ fsum, const float* __restrict__ fcnt,
                             const ull* __restrict__ pmin, const ull* __restrict__ pmax,
                             unsigned short* __restrict__ inbf) {
    long long gid = (long long)blockIdx.x * blockDim.x + threadIdx.x;
    if (gid >= (long long)N_NODES * K1_PAD) return;
    int n = (int)(gid / K1_PAD);
    int c = (int)(gid - (long long)n * K1_PAD);
    float v;
    if (c < 32) {
        v = x[(size_t)n * NC + c];
    } else if (c < 120) {
        unsigned e = (unsigned)(pmin[n] & 0xFFFFFFFFu);
        e = (e > (N_EDGES - 1u)) ? (N_EDGES - 1u) : e;
        v = ea[(size_t)e * EC_DIM + (c - 32)];
    } else if (c < 208) {
        unsigned e = 0xFFFFFFFFu - (unsigned)(pmax[n] & 0xFFFFFFFFu);
        e = (e > (N_EDGES - 1u)) ? (N_EDGES - 1u) : e;
        v = ea[(size_t)e * EC_DIM + (c - 120)];
    } else if (c < 296) {
        v = ord2f(umin[(size_t)n * EC_DIM + (c - 208)]);
    } else if (c < 384) {
        v = fsum[(size_t)n * EC_DIM + (c - 296)] / fmaxf(fcnt[n], 1.0f);
    } else if (c < 472) {
        v = ord2f(umax[(size_t)n * EC_DIM + (c - 384)]);
    } else {
        v = 0.0f;                 // K padding
    }
    inbf[gid] = f2bf(v);
}

// ---------------------------------------------------------------------------
// Kernel 4: transpose + convert weights to bf16 [N][K] so B fragments are
// contiguous 16B loads (matches the 16-bit WMMA operand layout).
// ---------------------------------------------------------------------------
__global__ void convert_weights(const float* __restrict__ W1, const float* __restrict__ W2,
                                const float* __restrict__ W3,
                                unsigned short* __restrict__ W1t,
                                unsigned short* __restrict__ W2t,
                                unsigned short* __restrict__ W3t) {
    int gid = blockIdx.x * blockDim.x + threadIdx.x;
    const int S1 = HC * K1_PAD;            // 256*480
    const int S2 = HC * HC;                // 256*256
    const int S3 = OC * HC;                // 32*256
    if (gid < S1) {
        int nn = gid / K1_PAD, k = gid - nn * K1_PAD;
        W1t[gid] = (k < IN_DIM) ? f2bf(W1[(size_t)k * HC + nn]) : (unsigned short)0;
    } else if (gid < S1 + S2) {
        int t = gid - S1;
        int nn = t / HC, k = t - nn * HC;
        W2t[t] = f2bf(W2[(size_t)k * HC + nn]);
    } else if (gid < S1 + S2 + S3) {
        int t = gid - S1 - S2;
        int nn = t / HC, k = t - nn * HC;
        W3t[t] = f2bf(W3[(size_t)k * OC + nn]);
    }
}

// ---------------------------------------------------------------------------
// Kernel 5: WMMA bf16 GEMM  Out[M,N] = relu?(A[M,K] * Bt[N,K]^T + bias)
// Each wave computes a 16 x (16*NB) strip: one A fragment is reused across
// NB accumulators. Per K-step, ALL 2+2*NB b128 loads are staged into distinct
// registers before the NB wmmas, so the scheduler can overlap the next step's
// loads with this step's wmmas (partial s_wait_loadcnt instead of full drain).
// A operand layout (16-bit, 16x32): lane m=l&15, halves at K in
// [koff, koff+8) and [koff+16, koff+24), koff = 8*(l>>4). B mirrors with N.
// ---------------------------------------------------------------------------
template<int K, int NB, bool RELU, typename OutT>
__global__ void gemm_wmma(const unsigned short* __restrict__ A,
                          const unsigned short* __restrict__ Bt,
                          const float* __restrict__ bias,
                          OutT* __restrict__ Out,
                          int Ngroups, long long totalUnits, int ldout) {
    const int wave = threadIdx.x >> 5;
    const int lane = threadIdx.x & 31;
    long long unit = (long long)blockIdx.x * (blockDim.x >> 5) + wave;
    if (unit >= totalUnits) return;            // wave-uniform: EXEC stays all-ones
    const int mt  = (int)(unit / Ngroups);
    const int ntg = (int)(unit - (long long)mt * Ngroups);

    const int half = lane >> 4;                // 0: lanes 0-15, 1: lanes 16-31
    const int idx  = lane & 15;
    const int koff = half * 8;

    const unsigned short* Ap = A  + (size_t)(mt * 16 + idx) * K + koff;
    const unsigned short* Bp = Bt + (size_t)(ntg * (16 * NB) + idx) * K + koff;

    v8f acc[NB];
#pragma unroll
    for (int nb = 0; nb < NB; ++nb) acc[nb] = (v8f){};

#pragma unroll
    for (int k0 = 0; k0 < K; k0 += 32) {
        // Stage all loads for this K-step into independent registers first.
        v8us alo = *(const v8us*)(Ap + k0);
        v8us ahi = *(const v8us*)(Ap + k0 + 16);
        v8us blo[NB], bhi[NB];
#pragma unroll
        for (int nb = 0; nb < NB; ++nb) {
            const unsigned short* Bpn = Bp + (size_t)nb * 16 * K;
            blo[nb] = *(const v8us*)(Bpn + k0);
            bhi[nb] = *(const v8us*)(Bpn + k0 + 16);
        }
        v16bf a = make_frag(alo, ahi);
#pragma unroll
        for (int nb = 0; nb < NB; ++nb) {
            v16bf b = make_frag(blo[nb], bhi[nb]);
            acc[nb] = __builtin_amdgcn_wmma_f32_16x16x32_bf16(
                /*neg_a=*/false, a, /*neg_b=*/false, b,
                /*c_mod=*/(short)0, acc[nb], /*reuse_a=*/false, /*reuse_b=*/false);
        }
    }

    // C/D layout: vgpr r, lane l -> m = r + 8*(l>=16), n = l&15
#pragma unroll
    for (int nb = 0; nb < NB; ++nb) {
        const int n = (ntg * NB + nb) * 16 + idx;
        const float bn = bias[n];
#pragma unroll
        for (int r = 0; r < 8; ++r) {
            float v = acc[nb][r] + bn;
            if (RELU) v = fmaxf(v, 0.0f);
            size_t off = (size_t)(mt * 16 + half * 8 + r) * ldout + n;
            store_out(&Out[off], v);
        }
    }
}

// ---------------------------------------------------------------------------
// Launcher
// ---------------------------------------------------------------------------
extern "C" void kernel_launch(void* const* d_in, const int* in_sizes, int n_in,
                              void* d_out, int out_size, void* d_ws, size_t ws_size,
                              hipStream_t stream) {
    const float* x    = (const float*)d_in[0];
    const int*   eidx = (const int*)  d_in[1];   // edge_index (2, E)
    const float* ea   = (const float*)d_in[2];   // edge_attr (E, 88)
    // d_in[3] = u (unused), d_in[4] = batch (unused)
    const float* W1 = (const float*)d_in[5];
    const float* b1 = (const float*)d_in[6];
    const float* W2 = (const float*)d_in[7];
    const float* b2 = (const float*)d_in[8];
    const float* W3 = (const float*)d_in[9];
    const float* b3 = (const float*)d_in[10];

    // Workspace carve-up (all 16B aligned)
    char* ws = (char*)d_ws;
    unsigned*       umin = (unsigned*)ws;       ws += (size_t)N_NODES * EC_DIM * 4;  // 17.6MB
    unsigned*       umax = (unsigned*)ws;       ws += (size_t)N_NODES * EC_DIM * 4;
    float*          fsum = (float*)ws;          ws += (size_t)N_NODES * EC_DIM * 4;
    float*          fcnt = (float*)ws;          ws += (size_t)N_NODES * 4;
    ull*            pmin = (ull*)ws;            ws += (size_t)N_NODES * 8;
    ull*            pmax = (ull*)ws;            ws += (size_t)N_NODES * 8;
    unsigned short* inbf = (unsigned short*)ws; ws += (size_t)N_NODES * K1_PAD * 2;  // 48MB
    unsigned short* h1   = (unsigned short*)ws; ws += (size_t)N_NODES * HC * 2;      // 25.6MB
    unsigned short* h2   = (unsigned short*)ws; ws += (size_t)N_NODES * HC * 2;      // 25.6MB
    unsigned short* W1t  = (unsigned short*)ws; ws += (size_t)HC * K1_PAD * 2;
    unsigned short* W2t  = (unsigned short*)ws; ws += (size_t)HC * HC * 2;
    unsigned short* W3t  = (unsigned short*)ws; ws += (size_t)OC * HC * 2;

    const int TB = 256;                        // 8 waves per block

    // 1) init
    {
        long long n = (long long)N_NODES * EC_DIM;
        init_agg<<<(unsigned)((n + TB - 1) / TB), TB, 0, stream>>>(umin, umax, fsum, fcnt, pmin, pmax);
    }
    // 2) edge aggregation
    {
        long long n = (long long)N_EDGES * EC_DIM;
        edge_agg<<<(unsigned)((n + TB - 1) / TB), TB, 0, stream>>>(ea, eidx, umin, umax, fsum, fcnt, pmin, pmax);
    }
    // 3) build input features (bf16)
    {
        long long n = (long long)N_NODES * K1_PAD;
        build_inputs<<<(unsigned)((n + TB - 1) / TB), TB, 0, stream>>>(x, ea, umin, umax, fsum, fcnt, pmin, pmax, inbf);
    }
    // 4) convert/transpose weights
    {
        int n = HC * K1_PAD + HC * HC + OC * HC;
        convert_weights<<<(n + TB - 1) / TB, TB, 0, stream>>>(W1, W2, W3, W1t, W2t, W3t);
    }
    // 5) GEMM chain (M = 50000 = 3125 tiles of 16, exact)
    const int Mtiles = N_NODES / 16;           // 3125
    {   // layer 1: [N,480] x [480,256] -> h1 bf16, ReLU; 16 N-tiles -> 4 groups of 4
        long long units = (long long)Mtiles * (HC / 64);   // 12500
        gemm_wmma<K1_PAD, 4, true, unsigned short>
            <<<(unsigned)((units + 7) / 8), TB, 0, stream>>>(inbf, W1t, b1, h1, HC / 64, units, HC);
    }
    {   // layer 2: [N,256] x [256,256] -> h2 bf16, ReLU
        long long units = (long long)Mtiles * (HC / 64);
        gemm_wmma<HC, 4, true, unsigned short>
            <<<(unsigned)((units + 7) / 8), TB, 0, stream>>>(h1, W2t, b2, h2, HC / 64, units, HC);
    }
    {   // layer 3: [N,256] x [256,32] -> out f32; 2 N-tiles -> 1 group of 2
        long long units = (long long)Mtiles * (OC / 32);   // 3125
        gemm_wmma<HC, 2, false, float>
            <<<(unsigned)((units + 7) / 8), TB, 0, stream>>>(h2, W3t, b3, (float*)d_out, OC / 32, units, OC);
    }
    (void)in_sizes; (void)n_in; (void)out_size; (void)ws_size;
}
